// Decoder_22273700397266
// MI455X (gfx1250) — matrix operand order repaired and checked by reading, not compiled, not run on hardware
//
#include <hip/hip_runtime.h>
#include <hip/hip_bf16.h>
#include <math.h>

typedef __attribute__((ext_vector_type(16))) _Float16 v16h;
typedef __attribute__((ext_vector_type(8)))  _Float16 v8h;
typedef __attribute__((ext_vector_type(8)))  float    v8f;

#define Bsz 64
#define Lsz 512
#define Hsz 1024
#define Dsz 2048
#define Msz (Bsz * Lsz)   // 32768

// ---------------------------------------------------------------------------
// s[b,h] = sum_l h_state[b,l,h]
__global__ __launch_bounds__(256) void k_sum_s(const float* __restrict__ h_state,
                                               float* __restrict__ s) {
    const int b = blockIdx.y;
    const int h = blockIdx.x * 256 + threadIdx.x;
    const float* p = h_state + (size_t)b * Lsz * Hsz + h;
    float acc = 0.0f;
    for (int l = 0; l < Lsz; ++l) acc += p[(size_t)l * Hsz];
    s[b * Hsz + h] = acc;
}

// sbias[b,e] = b_a[e] + b_s[e] + sum_h s[b,h]*W_s[e,h]
__global__ __launch_bounds__(256) void k_sbias(const float* __restrict__ s,
                                               const float* __restrict__ W_s,
                                               const float* __restrict__ b_a,
                                               const float* __restrict__ b_s,
                                               float* __restrict__ sbias) {
    __shared__ float sh[Hsz];
    const int b = blockIdx.y;
    const int t = threadIdx.x;
    const int e = blockIdx.x * 256 + t;
    for (int h = t; h < Hsz; h += 256) sh[h] = s[b * Hsz + h];
    __syncthreads();
    const float* w = W_s + (size_t)e * Hsz;
    float acc = 0.0f;
    for (int h = 0; h < Hsz; h += 4) {
        float4 wv = *(const float4*)(w + h);
        acc += wv.x * sh[h] + wv.y * sh[h + 1] + wv.z * sh[h + 2] + wv.w * sh[h + 3];
    }
    sbias[(size_t)b * Dsz + e] = acc + b_a[e] + b_s[e];
}

// a16[m, d] = f16( d<1024 ? h_state[m,d] : x[m,d-1024] ),  m = b*L + l
__global__ __launch_bounds__(256) void k_cvt_a(const float* __restrict__ h_state,
                                               const float* __restrict__ x,
                                               _Float16* __restrict__ a16) {
    const size_t i8 = ((size_t)blockIdx.x * 256 + threadIdx.x) * 8;
    const size_t m  = i8 >> 11;          // / 2048
    const int    d  = (int)(i8 & 2047);
    const float* src = (d < Hsz) ? (h_state + m * (size_t)Hsz + d)
                                 : (x       + m * (size_t)Hsz + (d - Hsz));
    float4 f0 = *(const float4*)(src);
    float4 f1 = *(const float4*)(src + 4);
    v8h o;
    o[0] = (_Float16)f0.x; o[1] = (_Float16)f0.y; o[2] = (_Float16)f0.z; o[3] = (_Float16)f0.w;
    o[4] = (_Float16)f1.x; o[5] = (_Float16)f1.y; o[6] = (_Float16)f1.z; o[7] = (_Float16)f1.w;
    *(v8h*)(a16 + i8) = o;
}

// w16 = f16(W_a), same row-major [e, d] layout
__global__ __launch_bounds__(256) void k_cvt_w(const float* __restrict__ W_a,
                                               _Float16* __restrict__ w16) {
    const size_t i8 = ((size_t)blockIdx.x * 256 + threadIdx.x) * 8;
    float4 f0 = *(const float4*)(W_a + i8);
    float4 f1 = *(const float4*)(W_a + i8 + 4);
    v8h o;
    o[0] = (_Float16)f0.x; o[1] = (_Float16)f0.y; o[2] = (_Float16)f0.z; o[3] = (_Float16)f0.w;
    o[4] = (_Float16)f1.x; o[5] = (_Float16)f1.y; o[6] = (_Float16)f1.z; o[7] = (_Float16)f1.w;
    *(v8h*)(w16 + i8) = o;
}

// ---------------------------------------------------------------------------
// Main fused GEMM: E[m,n] = exp(tanh( a[m,:] . W_a[n,:] + sbias[b(m), n] ))
// 128x128 tile, K-step 32, 8 wave32 waves (each 32M x 64N, 2x4 accumulators of
// v_wmma_f32_16x16x32_f16). Structurally double-buffered LDS (compile-time
// buffer indices) fed by GLOBAL_LOAD_ASYNC_TO_LDS (ASYNCcnt); two tiles per
// loop trip -> 16 WMMAs per trip, no branches or address math in steady state.
#define BK    32
#define LDT   40                         // padded LDS row stride (f16): 80B rows
#define TILEB (128 * LDT * 2)            // bytes per LDS buffer = 10240

// async 16B global->LDS copy; ISA adds INST_OFFSET to BOTH global & LDS address,
// so one base covers the thread's 32B segment via offset:0 / offset:16.
#define ASYNC_CP16(lds_off, gaddr)                                        \
    asm volatile("global_load_async_to_lds_b128 %0, %1, off"              \
                 :: "v"(lds_off), "v"(gaddr) : "memory")
#define ASYNC_CP16_O16(lds_off, gaddr)                                    \
    asm volatile("global_load_async_to_lds_b128 %0, %1, off offset:16"    \
                 :: "v"(lds_off), "v"(gaddr) : "memory")
#define WAIT_ASYNC0() asm volatile("s_wait_asynccnt 0x0" ::: "memory")

__global__ __launch_bounds__(256) void k_gemm(const _Float16* __restrict__ A,
                                              const _Float16* __restrict__ W,
                                              const float* __restrict__ sbias,
                                              _Float16* __restrict__ E) {
    __shared__ _Float16 As[2][128 * LDT];
    __shared__ _Float16 Bs[2][128 * LDT];

    const int t  = threadIdx.x;
    const int m0 = blockIdx.y * 128;
    const int n0 = blockIdx.x * 128;

    // cooperative tile load mapping: thread -> (row, 16-elem half-row)
    const int row  = t >> 1;
    const int half = (t & 1) * 16;
    const _Float16* gA = A + (size_t)(m0 + row) * Dsz + half;
    const _Float16* gB = W + (size_t)(n0 + row) * Dsz + half;
    // raw LDS byte offsets (low 32 bits of the generic pointer = LDS address)
    const unsigned ldsA = (unsigned)(uintptr_t)&As[0][row * LDT + half];
    const unsigned ldsB = (unsigned)(uintptr_t)&Bs[0][row * LDT + half];

    // wave/fragment mapping (ISA 7.12.2 layouts, wave32)
    const int lane = t & 31;
    const int wid  = t >> 5;
    const int wm   = (wid >> 1) * 32;   // 0,32,64,96
    const int wn   = (wid & 1) * 64;    // 0,64
    const int lr   = lane & 15;
    const int hiL  = lane >> 4;         // 0 | 1
    const int ak   = hiL * 8;           // A frag: K base 0 | 8 (pairs with +16)
    const int bk   = hiL * 8;           // B frag: K base 0 | 8 (pairs with +16)

    v8f acc[2][4];
#pragma unroll
    for (int i = 0; i < 2; ++i)
#pragma unroll
        for (int j = 0; j < 4; ++j)
            acc[i][j] = (v8f){0.f, 0.f, 0.f, 0.f, 0.f, 0.f, 0.f, 0.f};

    auto issue_tile = [&](int kk, int buf) {
        const unsigned la = ldsA + (unsigned)(buf * TILEB);
        const unsigned lb = ldsB + (unsigned)(buf * TILEB);
        const unsigned long long ga = (unsigned long long)(uintptr_t)(gA + kk);
        const unsigned long long gb = (unsigned long long)(uintptr_t)(gB + kk);
        ASYNC_CP16(la, ga);
        ASYNC_CP16_O16(la, ga);
        ASYNC_CP16(lb, gb);
        ASYNC_CP16_O16(lb, gb);
    };

    auto compute_tile = [&](const _Float16* Abuf, const _Float16* Bbuf) {
        // A fragments: lane<16 row=lr K{ak..+7, ak+16..+23}; lane>=16 K{8..15,24..31}
        v16h af[2];
#pragma unroll
        for (int mi = 0; mi < 2; ++mi) {
            const _Float16* p = &Abuf[(wm + mi * 16 + lr) * LDT + ak];
            v8h lo = *(const v8h*)(p);
            v8h hi = *(const v8h*)(p + 16);
            af[mi] = __builtin_shufflevector(lo, hi, 0, 1, 2, 3, 4, 5, 6, 7,
                                                     8, 9, 10, 11, 12, 13, 14, 15);
        }
        // all four B fragments up-front (distinct regs -> loads run ahead of MMAs)
        v16h bf[4];
#pragma unroll
        for (int ni = 0; ni < 4; ++ni) {
            const _Float16* p = &Bbuf[(wn + ni * 16 + lr) * LDT + bk];
            v8h lo = *(const v8h*)(p);
            v8h hi = *(const v8h*)(p + 16);
            bf[ni] = __builtin_shufflevector(lo, hi, 0, 1, 2, 3, 4, 5, 6, 7,
                                                     8, 9, 10, 11, 12, 13, 14, 15);
        }
#pragma unroll
        for (int ni = 0; ni < 4; ++ni) {
            acc[0][ni] = __builtin_amdgcn_wmma_f32_16x16x32_f16(
                false, af[0], false, bf[ni], (short)0, acc[0][ni], false, false);
            acc[1][ni] = __builtin_amdgcn_wmma_f32_16x16x32_f16(
                false, af[1], false, bf[ni], (short)0, acc[1][ni], false, false);
        }
    };

    // prologue: fill buffer 0 with tile k=0
    issue_tile(0, 0);
    WAIT_ASYNC0();
    __syncthreads();

    // steady state: two tiles per trip, compile-time buffer indices, no branches
    for (int k0 = 0; k0 < Dsz - 2 * BK; k0 += 2 * BK) {
        issue_tile(k0 + BK, 1);
        compute_tile(As[0], Bs[0]);
        WAIT_ASYNC0();
        __syncthreads();

        issue_tile(k0 + 2 * BK, 0);
        compute_tile(As[1], Bs[1]);
        WAIT_ASYNC0();
        __syncthreads();
    }
    // tail: last two tiles (k = Dsz-2*BK in buf0, k = Dsz-BK in buf1)
    issue_tile(Dsz - BK, 1);
    compute_tile(As[0], Bs[0]);
    WAIT_ASYNC0();
    __syncthreads();
    compute_tile(As[1], Bs[1]);

    // epilogue: E = exp(tanh(acc + bias));   (128 | 512 => whole block same batch)
    const int bb = m0 >> 9;
    const float* sb = sbias + (size_t)bb * Dsz;
#pragma unroll
    for (int mi = 0; mi < 2; ++mi) {
        const int mbase = m0 + wm + mi * 16 + hiL * 8;
#pragma unroll
        for (int ni = 0; ni < 4; ++ni) {
            const int ncol = n0 + wn + ni * 16 + lr;
            const float bias = sb[ncol];
#pragma unroll
            for (int v = 0; v < 8; ++v) {
                float val = __expf(tanhf(acc[mi][ni][v] + bias));
                E[(size_t)(mbase + v) * Dsz + ncol] = (_Float16)val;
            }
        }
    }
}

// ---------------------------------------------------------------------------
// rZ[m] = 1 / sum_e E[m,e]
__global__ __launch_bounds__(256) void k_rowsum(const _Float16* __restrict__ E,
                                                float* __restrict__ rZ) {
    __shared__ float red[256];
    const int m = blockIdx.x;
    const int t = threadIdx.x;
    const _Float16* rowp = E + (size_t)m * Dsz;
    v8h v = *(const v8h*)(rowp + t * 8);   // 256 threads * 8 = 2048
    float s = 0.0f;
#pragma unroll
    for (int i = 0; i < 8; ++i) s += (float)v[i];
    red[t] = s;
    __syncthreads();
    for (int off = 128; off > 0; off >>= 1) {
        if (t < off) red[t] += red[t + off];
        __syncthreads();
    }
    if (t == 0) rZ[m] = 1.0f / red[0];
}

// out[b,e] = a[b,trig,e] * ( sum_{l: mask!=0} E[m,e]*rZ[m] + n_masked/2048 )
__global__ __launch_bounds__(256) void k_out(const _Float16* __restrict__ E,
                                             const float* __restrict__ rZ,
                                             const int* __restrict__ mask,
                                             const int* __restrict__ trigger,
                                             const float* __restrict__ h_state,
                                             const float* __restrict__ x,
                                             float* __restrict__ out) {
    __shared__ float w[Lsz];
    __shared__ float redc[256];
    const int b = blockIdx.y;
    const int t = threadIdx.x;
    const int e = blockIdx.x * 256 + t;

    int zc = 0;
    for (int l = t; l < Lsz; l += 256) {
        const int mk = mask[b * Lsz + l];
        w[l] = mk ? rZ[b * Lsz + l] : 0.0f;
        zc += (mk == 0);
    }
    redc[t] = (float)zc;
    __syncthreads();
    for (int off = 128; off > 0; off >>= 1) {
        if (t < off) redc[t] += redc[t + off];
        __syncthreads();
    }
    const float c0 = redc[0];

    float acc = 0.0f;
    const _Float16* Eb = E + (size_t)b * Lsz * Dsz + e;
    for (int l = 0; l < Lsz; ++l)
        acc += (float)Eb[(size_t)l * Dsz] * w[l];
    acc += c0 * (1.0f / (float)Dsz);

    const int trig = trigger[b];
    const float ta = (e < Hsz)
        ? h_state[((size_t)b * Lsz + trig) * Hsz + e]
        : x[((size_t)b * Lsz + trig) * Hsz + (e - Hsz)];
    out[(size_t)b * Dsz + e] = ta * acc;
}

// ---------------------------------------------------------------------------
extern "C" void kernel_launch(void* const* d_in, const int* in_sizes, int n_in,
                              void* d_out, int out_size, void* d_ws, size_t ws_size,
                              hipStream_t stream) {
    (void)in_sizes; (void)n_in; (void)out_size; (void)ws_size;
    const float* h_state = (const float*)d_in[0];
    const float* x       = (const float*)d_in[1];
    const float* W_a     = (const float*)d_in[2];
    const float* b_a     = (const float*)d_in[3];
    const float* W_s     = (const float*)d_in[4];
    const float* b_s     = (const float*)d_in[5];
    const int*   trigger = (const int*)d_in[6];
    const int*   mask    = (const int*)d_in[7];
    float* out = (float*)d_out;

    char* ws = (char*)d_ws;
    size_t off = 0;
    _Float16* a16 = (_Float16*)(ws + off); off += (size_t)Msz * Dsz * 2;   // 134 MB
    _Float16* E16 = (_Float16*)(ws + off); off += (size_t)Msz * Dsz * 2;   // 134 MB
    _Float16* w16 = (_Float16*)(ws + off); off += (size_t)Dsz * Dsz * 2;   //   8 MB
    float*    s    = (float*)(ws + off);   off += (size_t)Bsz * Hsz * 4;
    float*    sbias= (float*)(ws + off);   off += (size_t)Bsz * Dsz * 4;
    float*    rZ   = (float*)(ws + off);   off += (size_t)Msz * 4;

    k_sum_s<<<dim3(Hsz / 256, Bsz), 256, 0, stream>>>(h_state, s);
    k_sbias<<<dim3(Dsz / 256, Bsz), 256, 0, stream>>>(s, W_s, b_a, b_s, sbias);
    k_cvt_a<<<(size_t)Msz * Dsz / (256 * 8), 256, 0, stream>>>(h_state, x, a16);
    k_cvt_w<<<(size_t)Dsz * Dsz / (256 * 8), 256, 0, stream>>>(W_a, w16);
    k_gemm<<<dim3(Dsz / 128, Msz / 128), 256, 0, stream>>>(a16, w16, sbias, E16);
    k_rowsum<<<Msz, 256, 0, stream>>>(E16, rZ);
    k_out<<<dim3(Dsz / 256, Bsz), 256, 0, stream>>>(E16, rZ, mask, trigger,
                                                    h_state, x, out);
}
